// FractalMultiplier_31825707663735
// MI455X (gfx1250) — matrix operand order repaired and checked by reading, not compiled, not run on hardware
//
#include <hip/hip_runtime.h>

// ---------------------------------------------------------------------------
// FractalMultiplier forward on gfx1250 (MI455X), wave32 + WMMA.
//
// Key simplifications derived from the reference:
//  * acc/term rows are always rows of the 2-entry table E, so the input
//    projection x@W reduces to 4 precomputed 768-vectors (P0,P1,Q0+b,Q1+b).
//  * argmax over V=2 is a sign test of two dots against W_o.
//  * Only h@U ([16,256]x[256,768] per block-cell) remains on the 1024-long
//    serial path -> done with v_wmma_f32_16x16x32_f16, U held in VGPRs.
//  * All 8 A-fragments are preloaded per cell so the 24 WMMAs issue
//    back-to-back behind one s_wait_dscnt (latency-bound critical path).
// ---------------------------------------------------------------------------

typedef _Float16 v16h __attribute__((ext_vector_type(16)));
typedef _Float16 v8h  __attribute__((ext_vector_type(8)));
typedef float    v8f  __attribute__((ext_vector_type(8)));

#define NB   16          // blocks (256 batch rows / 16 per block)
#define BT   16          // batch rows per block (one WMMA M-tile)
#define SQ   32          // sequence length / outer steps
#define DD   128         // embed dim
#define HH   256         // GRU hidden
#define GG   768         // 3*H
#define TPB  512         // 16 waves of 32
#define HP   (HH + 8)    // padded h row (528 B, 16B-aligned stride)

__device__ __forceinline__ v8f wmma16(v16h a, v16h b, v8f c) {
  // D = A(16x32 f16) * B(32x16 f16) + C(16x16 f32)
  return __builtin_amdgcn_wmma_f32_16x16x32_f16(
      /*neg_a=*/false, a, /*neg_b=*/false, b,
      /*c_mod=*/(short)0, c, /*reuse_a=*/false, /*reuse_b=*/false);
}

__device__ __forceinline__ float sigmoid_(float x) {
  return 1.0f / (1.0f + __expf(-x));
}
__device__ __forceinline__ float tanh_(float x) {
  float xc = fminf(fmaxf(x, -15.0f), 15.0f);
  float e  = __expf(2.0f * xc);
  return (e - 1.0f) / (e + 1.0f);
}

extern "C" __global__ void __launch_bounds__(TPB, 1)
fractal_gru_wmma(const int* __restrict__ a_seq, const int* __restrict__ b_seq,
                 const float* __restrict__ E,  const float* __restrict__ W,
                 const float* __restrict__ U,  const float* __restrict__ bias,
                 const float* __restrict__ W_o, const float* __restrict__ b_o,
                 float* __restrict__ out)
{
  __shared__ __align__(16) _Float16 h16[BT][HP];   // GRU state, f16, A-matrix source
  __shared__ float Ptab[4][GG];                    // P0,P1,Q0+b,Q1+b
  __shared__ float Wo[HH][2];
  __shared__ float bo2[2];
  __shared__ float part0[TPB], part1[TPB];
  __shared__ unsigned char abits[BT][SQ], bbits[BT][SQ], tacc[BT][SQ];

  const int t    = threadIdx.x;
  const int w    = t >> 5;     // wave id 0..15
  const int l    = t & 31;     // lane
  const int half = l >> 4;     // lane half (K-half select)
  const int ln   = l & 15;
  const int b0   = blockIdx.x * BT;

  // ---------------- one-time: xg lookup tables (kills the x@W GEMM) --------
  for (int e = t; e < 4 * GG; e += TPB) {
    const int tab = e / GG, col = e - tab * GG;
    const float* er = E + (tab & 1) * DD;              // E[0] or E[1]
    const float* wb = W + (tab >> 1) * DD * GG + col;  // W top/bottom half
    float s = (tab >= 2) ? bias[col] : 0.0f;           // fold b into Q tables
    for (int d = 0; d < DD; ++d) s = fmaf(er[d], wb[d * GG], s);
    Ptab[tab][col] = s;
  }
  for (int e = t; e < HH * 2; e += TPB) Wo[e >> 1][e & 1] = W_o[e];
  if (t < 2) bo2[t] = b_o[t];
  for (int e = t; e < BT * SQ; e += TPB) {
    const int b = e >> 5, s = e & 31;
    abits[b][s] = (unsigned char)a_seq[(b0 + b) * SQ + s];
    bbits[b][s] = (unsigned char)b_seq[(b0 + b) * SQ + s];
    tacc[b][s]  = 0;                                   // acc0 = E[0] everywhere
  }

  // ---------------- one-time: U -> f16 B-fragments, pinned in VGPRs --------
  // Wave w owns N-tiles {w, w+16, w+32}  (columns j, 256+j, 512+j of hg).
// B 32x16 striping (matches A): element j<8 is K = 8*half + j, element
// j>=8 is K = 16 + 8*half + (j-8), for this lane's column.
  v16h Bfrag[3][8];
  #pragma unroll
  for (int c = 0; c < 3; ++c) {
    const int ncol = (w + 16 * c) * 16 + ln;
    #pragma unroll
    for (int k = 0; k < 8; ++k) {
      const int kb = k * 32 + half * 8;
      v16h bf;
      #pragma unroll
      for (int j = 0; j < 16; ++j)
        bf[j] = (_Float16)U[(kb + j + (j >= 8 ? 8 : 0)) * GG + ncol];
      Bfrag[c][k] = bf;
    }
  }
  __syncthreads();

  const int jl = w * 16 + ln;   // hidden column this lane owns in gate phase

  #pragma unroll 1
  for (int i = 0; i < SQ; ++i) {                 // outer (serial: argmax feedback)
    for (int e = t; e < BT * HP; e += TPB)       // h0 = 0 each outer step
      ((_Float16*)h16)[e] = (_Float16)0.0f;
    __syncthreads();

    #pragma unroll 1
    for (int s = 0; s < SQ; ++s) {               // GRU scan (serial)
      // ---- hg = h @ U via WMMA; f32 accumulate -------------------------
      v8f accZ, accR, accN;
      #pragma unroll
      for (int e = 0; e < 8; ++e) { accZ[e] = 0.f; accR[e] = 0.f; accN[e] = 0.f; }

      // Preload all 8 A-fragments (rows = ln; K chunks [half*8,+8) and
      // [16+half*8,+8) in each 32-wide K step -> 16x ds_load_b128), then
      // issue the 24 WMMAs back-to-back behind a single LDS wait.
      union AF { v16h v; v8h h2[2]; };
      AF a[8];
      #pragma unroll
      for (int k = 0; k < 8; ++k) {
        const int kb = k * 32 + half * 8;
        a[k].h2[0] = *(const v8h*)&h16[ln][kb];
        a[k].h2[1] = *(const v8h*)&h16[ln][kb + 16];
      }
      #pragma unroll
      for (int k = 0; k < 8; ++k) {
        accZ = wmma16(a[k].v, Bfrag[0][k], accZ);
        accR = wmma16(a[k].v, Bfrag[1][k], accR);
        accN = wmma16(a[k].v, Bfrag[2][k], accN);
      }
      __syncthreads();   // all waves done reading h16 before overwrite

      // ---- GRU gates, intra-wave (z/r/n columns co-resident) -----------
      // C/D layout: element v -> row M = v + 8*half, col = ln.
      #pragma unroll
      for (int v = 0; v < 8; ++v) {
        const int br = half * 8 + v;
        const int ta = tacc[br][s];
        const int sa = s - i;
        const int tt = (bbits[br][i] && sa >= 0) ? abits[br][sa] : 0;
        const float xz = Ptab[ta][jl]          + Ptab[2 + tt][jl];
        const float xr = Ptab[ta][HH + jl]     + Ptab[2 + tt][HH + jl];
        const float xn = Ptab[ta][2 * HH + jl] + Ptab[2 + tt][2 * HH + jl];
        const float z  = sigmoid_(xz + accZ[v]);
        const float r  = sigmoid_(xr + accR[v]);
        const float n  = tanh_(xn + r * accN[v]);
        const float ho = (float)h16[br][jl];
        h16[br][jl] = (_Float16)((1.0f - z) * ho + z * n);
      }
      __syncthreads();

      // ---- logits / hard token: two dots vs W_o ------------------------
      {
        const int db = t >> 5;      // batch row 0..15
        const int dc = t & 31;      // 8-wide chunk 0..31
        float s0 = 0.f, s1 = 0.f;
        #pragma unroll
        for (int kk = dc * 8; kk < dc * 8 + 8; ++kk) {
          const float hv = (float)h16[db][kk];
          s0 = fmaf(hv, Wo[kk][0], s0);
          s1 = fmaf(hv, Wo[kk][1], s1);
        }
        part0[t] = s0; part1[t] = s1;
      }
      __syncthreads();
      if (t < BT) {
        float l0 = bo2[0], l1 = bo2[1];
        for (int c = 0; c < 32; ++c) { l0 += part0[t * 32 + c]; l1 += part1[t * 32 + c]; }
        tacc[t][s] = (l1 > l0) ? (unsigned char)1 : (unsigned char)0;  // argmax, tie->0
        if (i == SQ - 1) {
          const int o = ((b0 + t) * SQ + s) * 2;
          out[o] = l0; out[o + 1] = l1;
        }
      }
      __syncthreads();
    }
  }
}

extern "C" void kernel_launch(void* const* d_in, const int* in_sizes, int n_in,
                              void* d_out, int out_size, void* d_ws, size_t ws_size,
                              hipStream_t stream) {
  const int*   a_seq = (const int*)d_in[0];
  const int*   b_seq = (const int*)d_in[1];
  const float* E     = (const float*)d_in[2];
  const float* W     = (const float*)d_in[3];
  const float* U     = (const float*)d_in[4];
  const float* bias  = (const float*)d_in[5];
  const float* W_o   = (const float*)d_in[6];
  const float* b_o   = (const float*)d_in[7];
  float* out = (float*)d_out;

  fractal_gru_wmma<<<NB, TPB, 0, stream>>>(a_seq, b_seq, E, W, U, bias, W_o, b_o, out);

  (void)in_sizes; (void)n_in; (void)out_size; (void)d_ws; (void)ws_size;
}
